// MambaLayer_17695265260108
// MI455X (gfx1250) — matrix operand order repaired and checked by reading, not compiled, not run on hardware
//
#include <hip/hip_runtime.h>

// ---------------------------------------------------------------------------
// MI455X (gfx1250) Mamba layer: bf16 WMMA GEMMs (TDM-staged A tiles) + scan
// B=4, L=2048, D=2048, E=4096, N=16, DT=128, DC=4
// ---------------------------------------------------------------------------

typedef __attribute__((ext_vector_type(16))) __bf16        v16bf;
typedef __attribute__((ext_vector_type(8)))  float         v8f;
typedef __attribute__((ext_vector_type(4)))  unsigned int  v4u;
typedef __attribute__((ext_vector_type(8)))  unsigned int  v8u;
typedef __attribute__((ext_vector_type(4)))  int           v4i_;
typedef __attribute__((ext_vector_type(8)))  int           v8i_;

union Frag32 { v16bf v; uint4 q[2]; };

__device__ __forceinline__ unsigned short f2bf(float f) {
    unsigned int u = __builtin_bit_cast(unsigned int, f);
    unsigned int r = u + 0x7FFFu + ((u >> 16) & 1u);   // round-to-nearest-even
    return (unsigned short)(r >> 16);
}
__device__ __forceinline__ float bf2f(unsigned short h) {
    unsigned int u = ((unsigned int)h) << 16;
    return __builtin_bit_cast(float, u);
}

// ---------------------------------------------------------------------------
// TDM: issue tensor_load_to_lds of a [tile_rows x 32] bf16 tile of A into LDS.
// D# layout per CDNA5 ISA (group0 128b, group1 256b).
// ---------------------------------------------------------------------------
__device__ __forceinline__ void tdm_load_a_tile(const __bf16* A, unsigned long long row0,
                                                int K, int k0, unsigned int lds_addr,
                                                v8u g1u) {
    unsigned long long ga = (unsigned long long)(size_t)A + (row0 * (unsigned long long)K
                            + (unsigned long long)k0) * 2ull;
    v4u g0;
    g0[0] = 1u;                                        // count=1, user mode
    g0[1] = lds_addr;                                  // LDS byte address
    g0[2] = (unsigned int)ga;                          // global_addr[31:0]
    g0[3] = (unsigned int)((ga >> 32) & 0x01FFFFFFull) // global_addr[56:32]
          | (2u << 30);                                // type = 2 ("image")
#if __has_builtin(__builtin_amdgcn_tensor_load_to_lds)
    v8i_ g1i = __builtin_bit_cast(v8i_, g1u);
    v4i_ z4 = {};
#  if __clang_major__ >= 23
    v8i_ z8 = {};
    __builtin_amdgcn_tensor_load_to_lds(g0, g1i, z4, z4, z8, 0);
#  else
    __builtin_amdgcn_tensor_load_to_lds(g0, g1i, z4, z4, 0);
#  endif
#else
    asm volatile("tensor_load_to_lds %0, %1" :: "s"(g0), "s"(g1u) : "memory");
#endif
}

// Build loop-invariant D# group1: 2-D tensor [M x K] bf16, tile 32(K) x 128(rows)
__device__ __forceinline__ v8u tdm_make_g1(int M, int K) {
    v8u g1;
    g1[0] = (1u << 16);                                    // data_size = 2 bytes
    g1[1] = ((unsigned int)K & 0xFFFFu) << 16;             // tensor_dim0[15:0]
    g1[2] = ((unsigned int)K >> 16)                        // tensor_dim0[31:16]
          | (((unsigned int)M & 0xFFFFu) << 16);           // tensor_dim1[15:0]
    g1[3] = ((unsigned int)M >> 16) | (32u << 16);         // tensor_dim1[31:16], tile_dim0=32
    g1[4] = 128u;                                          // tile_dim1=128, tile_dim2=0
    g1[5] = (unsigned int)K;                               // tensor_dim0_stride[31:0]
    g1[6] = 0u;                                            // stride0 hi, stride1 lo
    g1[7] = 0u;
    return g1;
}

// ---------------------------------------------------------------------------
// f32 -> bf16 conversion (weights, one-time per launch)
// ---------------------------------------------------------------------------
__global__ void f32_to_bf16_kernel(const float* __restrict__ in,
                                   unsigned short* __restrict__ out, size_t n) {
    size_t i = (size_t)blockIdx.x * blockDim.x + threadIdx.x;
    size_t stride = (size_t)gridDim.x * blockDim.x;
    for (; i < n; i += stride) out[i] = f2bf(in[i]);
}

// ---------------------------------------------------------------------------
// RMSNorm: one block per row of D, LDS tree reduction, bf16 output
// ---------------------------------------------------------------------------
__global__ __launch_bounds__(256)
void rmsnorm_kernel(const float* __restrict__ x, const float* __restrict__ w,
                    unsigned short* __restrict__ xn, int D) {
    __shared__ float red[256];
    const int row = blockIdx.x;
    const float* xr = x + (size_t)row * D;
    float s = 0.f;
    for (int d = threadIdx.x; d < D; d += 256) { float v = xr[d]; s += v * v; }
    red[threadIdx.x] = s;
    __syncthreads();
    for (int off = 128; off > 0; off >>= 1) {
        if (threadIdx.x < off) red[threadIdx.x] += red[threadIdx.x + off];
        __syncthreads();
    }
    const float scale = rsqrtf(red[0] / (float)D + 1e-5f);
    for (int d = threadIdx.x; d < D; d += 256)
        xn[(size_t)row * D + d] = f2bf(xr[d] * scale * w[d]);
}

// ---------------------------------------------------------------------------
// GEMM: Out[M,N] = A[M,K](bf16) * W[N,K](bf16)^T
//   mode 0: store bf16   mode 1: store f32   mode 2: f32 softplus(acc+bias[n])
// Block: 256 threads (8 waves), tile 128M x 128N; wave tile 32M x 64N
//   => 8 v_wmma_f32_16x16x32_bf16 per 32-wide k-step per wave.
// A block tile (128x32 bf16, 8KB) is DMA'd to LDS by the Tensor Data Mover,
// double-buffered and pipelined with s_wait_tensorcnt; B rows are loaded
// directly from global (clamped pointers, N % 16 == 0 so fragments are
// uniformly valid/invalid and invalid ones are simply never stored).
// A/B fragments use contiguous k (lane half h holds k = 16h..16h+15); C per ISA.
// ---------------------------------------------------------------------------
__global__ __launch_bounds__(256)
void gemm_bf16_kernel(const __bf16* __restrict__ A, const __bf16* __restrict__ W,
                      void* __restrict__ Out, const float* __restrict__ bias,
                      int M, int N, int K, int mode) {
    __shared__ uint4 abuf[2][512];          // 2 x 8KB A tiles (128 rows x 64B)

    const int lane = threadIdx.x & 31;
    const int wave = threadIdx.x >> 5;
    const int wr = wave >> 1;               // 0..3 along M
    const int wc = wave & 1;                // 0..1 along N
    const int hi = lane >> 4;               // half-wave select
    const int ln = lane & 15;

    const int mblk = blockIdx.y * 128;
    const int n0 = blockIdx.x * 128 + wc * 64;

    // B row pointers, clamped (fragments are uniformly valid since N%16==0)
    const __bf16* brow[4];
#pragma unroll
    for (int t = 0; t < 4; ++t) {
        int n = n0 + t * 16 + ln;
        brow[t] = W + (size_t)(n < N ? n : 0) * K + hi * 16;
    }

    v8f zf = {};
    v8f acc[2][4];
#pragma unroll
    for (int mt = 0; mt < 2; ++mt)
#pragma unroll
        for (int t = 0; t < 4; ++t) acc[mt][t] = zf;

    const v8u g1 = tdm_make_g1(M, K);
    const unsigned int lds0 = (unsigned int)(size_t)&abuf[0][0];
    const unsigned int lds1 = (unsigned int)(size_t)&abuf[1][0];

    const int nk = K / 32;
    if (wave == 0) {
        tdm_load_a_tile(A, (unsigned long long)mblk, K, 0, lds0, g1);
        if (nk > 1) tdm_load_a_tile(A, (unsigned long long)mblk, K, 32, lds1, g1);
    }

    for (int i = 0; i < nk; ++i) {
        if (wave == 0) {
            if (i + 1 < nk) __builtin_amdgcn_s_wait_tensorcnt(1);
            else            __builtin_amdgcn_s_wait_tensorcnt(0);
        }
        __syncthreads();                    // tile i visible in abuf[i&1]

        const int k0 = i * 32;
        const uint4* ab = &abuf[i & 1][0];

        // B fragments (global, contiguous 32B per lane)
        uint4 b0[4], b1[4];
#pragma unroll
        for (int t = 0; t < 4; ++t) {
            const uint4* bp = (const uint4*)(brow[t] + k0);
            b0[t] = bp[0];
            b1[t] = bp[1];
        }
        __builtin_prefetch(brow[0] + k0 + 512, 0, 1);   // global_prefetch_b8

        // A fragments from LDS (ds_load_b128)
        Frag32 af[2];
#pragma unroll
        for (int mt = 0; mt < 2; ++mt) {
            int row = wr * 32 + mt * 16 + ln;           // 0..127
            const uint4* ap = ab + row * 4;             // 64B per row
            af[mt].q[0] = ap[2 * hi];
            af[mt].q[1] = ap[2 * hi + 1];
        }

#pragma unroll
        for (int mt = 0; mt < 2; ++mt) {
#pragma unroll
            for (int t = 0; t < 4; ++t) {
                Frag32 bf;
                bf.q[0] = b0[t];
                bf.q[1] = b1[t];
                acc[mt][t] = __builtin_amdgcn_wmma_f32_16x16x32_bf16(
                    false, af[mt].v, false, bf.v, (short)0, acc[mt][t], false, false);
            }
        }

        __syncthreads();                    // everyone done reading abuf[i&1]
        if (wave == 0 && i + 2 < nk)
            tdm_load_a_tile(A, (unsigned long long)mblk, K, (i + 2) * 32,
                            (i & 1) ? lds1 : lds0, g1);
    }

    // C layout: VGPR i -> row base + hi*8 + i, col = n0 + t*16 + ln
#pragma unroll
    for (int mt = 0; mt < 2; ++mt) {
#pragma unroll
        for (int t = 0; t < 4; ++t) {
            int col = n0 + t * 16 + ln;
            if (col >= N) continue;
            int rb = mblk + wr * 32 + mt * 16 + hi * 8;
#pragma unroll
            for (int i = 0; i < 8; ++i) {
                float v = acc[mt][t][i];
                size_t idx = (size_t)(rb + i) * N + col;
                if (mode == 0) {
                    ((unsigned short*)Out)[idx] = f2bf(v);
                } else if (mode == 1) {
                    ((float*)Out)[idx] = v;
                } else {
                    float z = v + bias[col];
                    ((float*)Out)[idx] = (z > 20.f) ? z : log1pf(__expf(z));
                }
            }
        }
    }
}

// ---------------------------------------------------------------------------
// Causal depthwise conv1d (DC=4) + bias + SiLU, bf16 in/out
// ---------------------------------------------------------------------------
__global__ __launch_bounds__(256)
void conv_silu_kernel(const unsigned short* __restrict__ upre,
                      const float* __restrict__ cw, const float* __restrict__ cb,
                      unsigned short* __restrict__ u, int Lx, int E, size_t total) {
    size_t idx = (size_t)blockIdx.x * blockDim.x + threadIdx.x;
    if (idx >= total) return;
    int e = (int)(idx % E);
    size_t bl = idx / E;             // b*L + l
    int l = (int)(bl % Lx);
    float acc = cb[e];
#pragma unroll
    for (int j = 0; j < 4; ++j) {
        int back = 3 - j;
        if (l >= back)
            acc += bf2f(upre[(bl - (size_t)back) * E + e]) * cw[e * 4 + j];
    }
    float s = acc / (1.f + __expf(-acc));   // silu
    u[idx] = f2bf(s);
}

// ---------------------------------------------------------------------------
// Selective scan (serial in L, parallel over B*E), fused output gating:
//   h[n] = exp(delta*A[n])*h[n] + delta*u*B[n];  y = sum_n h[n]*C[n]
//   out  = (y + u*W_D) * silu(skip)  -> bf16 for final GEMM
// ---------------------------------------------------------------------------
__global__ __launch_bounds__(256)
void scan_kernel(const float* __restrict__ delta, const unsigned short* __restrict__ u,
                 const float* __restrict__ Bt, const float* __restrict__ Ct,
                 const float* __restrict__ A_log, const float* __restrict__ W_D,
                 const unsigned short* __restrict__ skip,
                 unsigned short* __restrict__ yb, int Lx, int E) {
    const int e = blockIdx.x * 256 + threadIdx.x;
    const int b = blockIdx.y;
    float a[16], h[16];
#pragma unroll
    for (int n = 0; n < 16; ++n) {
        a[n] = -__expf(A_log[(size_t)e * 16 + n]);
        h[n] = 0.f;
    }
    const float wd = W_D[e];
    for (int l = 0; l < Lx; ++l) {
        size_t be = ((size_t)b * Lx + l) * E + e;
        size_t bn = ((size_t)b * Lx + l) * 16;
        float d  = delta[be];
        float uu = bf2f(u[be]);
        float du = d * uu;
        float y = 0.f;
#pragma unroll
        for (int n = 0; n < 16; ++n) {
            h[n] = __expf(d * a[n]) * h[n] + du * Bt[bn + n];
            y += h[n] * Ct[bn + n];
        }
        float sk = bf2f(skip[be]);
        float g  = sk / (1.f + __expf(-sk));     // silu(skip)
        yb[be] = f2bf((y + uu * wd) * g);
    }
}

// ---------------------------------------------------------------------------
// Host-side orchestration
// ---------------------------------------------------------------------------
extern "C" void kernel_launch(void* const* d_in, const int* in_sizes, int n_in,
                              void* d_out, int out_size, void* d_ws, size_t ws_size,
                              hipStream_t stream) {
    (void)in_sizes; (void)n_in; (void)out_size; (void)ws_size;

    const int Bc = 4, Lc = 2048, Dc = 2048, Ec = 4096, Nc = 16, DTc = 128;
    const int Mr = Bc * Lc; // 8192 rows

    const float* x      = (const float*)d_in[0];
    const float* norm_w = (const float*)d_in[1];
    const float* W_skip = (const float*)d_in[2];
    const float* W_in   = (const float*)d_in[3];
    const float* conv_w = (const float*)d_in[4];
    const float* conv_b = (const float*)d_in[5];
    const float* Wd1    = (const float*)d_in[6];
    const float* Wd2    = (const float*)d_in[7];
    const float* bd2    = (const float*)d_in[8];
    const float* WB     = (const float*)d_in[9];
    const float* WC     = (const float*)d_in[10];
    const float* A_log  = (const float*)d_in[11];
    const float* W_D    = (const float*)d_in[12];
    const float* W_out  = (const float*)d_in[13];

    char* ws = (char*)d_ws;
    size_t off = 0;
    auto alloc = [&](size_t bytes) -> void* {
        void* p = ws + off;
        off = (off + bytes + 255) & ~(size_t)255;
        return p;
    };

    unsigned short* xn    = (unsigned short*)alloc((size_t)Mr * Dc * 2);
    unsigned short* Wskb  = (unsigned short*)alloc((size_t)Ec * Dc * 2);
    unsigned short* Winb  = (unsigned short*)alloc((size_t)Ec * Dc * 2);
    unsigned short* Wd1b  = (unsigned short*)alloc((size_t)DTc * Ec * 2);
    unsigned short* Wd2b  = (unsigned short*)alloc((size_t)Ec * DTc * 2);
    unsigned short* WBb   = (unsigned short*)alloc((size_t)Nc * Ec * 2);
    unsigned short* WCb   = (unsigned short*)alloc((size_t)Nc * Ec * 2);
    unsigned short* Woub  = (unsigned short*)alloc((size_t)Dc * Ec * 2);
    unsigned short* skipb = (unsigned short*)alloc((size_t)Mr * Ec * 2);
    unsigned short* upreb = (unsigned short*)alloc((size_t)Mr * Ec * 2);
    unsigned short* ub    = (unsigned short*)alloc((size_t)Mr * Ec * 2);
    unsigned short* t1b   = (unsigned short*)alloc((size_t)Mr * DTc * 2);
    float*          dltf  = (float*)alloc((size_t)Mr * Ec * 4);
    float*          Btf   = (float*)alloc((size_t)Mr * Nc * 4);
    float*          Ctf   = (float*)alloc((size_t)Mr * Nc * 4);
    unsigned short* ybb   = (unsigned short*)alloc((size_t)Mr * Ec * 2);

    auto cvt = [&](const float* src, unsigned short* dst, size_t n) {
        int blocks = (int)((n + 2047) / 2048);
        if (blocks < 1) blocks = 1;
        f32_to_bf16_kernel<<<blocks, 256, 0, stream>>>(src, dst, n);
    };
    cvt(W_skip, Wskb, (size_t)Ec * Dc);
    cvt(W_in,   Winb, (size_t)Ec * Dc);
    cvt(Wd1,    Wd1b, (size_t)DTc * Ec);
    cvt(Wd2,    Wd2b, (size_t)Ec * DTc);
    cvt(WB,     WBb,  (size_t)Nc * Ec);
    cvt(WC,     WCb,  (size_t)Nc * Ec);
    cvt(W_out,  Woub, (size_t)Dc * Ec);

    // 1) RMSNorm -> xn (bf16)
    rmsnorm_kernel<<<Mr, 256, 0, stream>>>(x, norm_w, xn, Dc);

    auto gemm = [&](const unsigned short* A, const unsigned short* W, void* Out,
                    const float* bias, int M, int N, int K, int mode) {
        dim3 grid((N + 127) / 128, M / 128);
        gemm_bf16_kernel<<<grid, 256, 0, stream>>>(
            (const __bf16*)A, (const __bf16*)W, Out, bias, M, N, K, mode);
    };

    // 2) skip = xn @ W_skip^T ; u_pre = xn @ W_in^T   (bf16 outputs)
    gemm(xn, Wskb, skipb, nullptr, Mr, Ec, Dc, 0);
    gemm(xn, Winb, upreb, nullptr, Mr, Ec, Dc, 0);

    // 3) causal depthwise conv + SiLU -> u (bf16)
    {
        size_t total = (size_t)Mr * Ec;
        int blocks = (int)((total + 255) / 256);
        conv_silu_kernel<<<blocks, 256, 0, stream>>>(upreb, conv_w, conv_b, ub,
                                                     Lc, Ec, total);
    }

    // 4) delta = softplus(u @ Wd1^T @ Wd2^T + bd2)
    gemm(ub,  Wd1b, t1b,  nullptr, Mr, DTc, Ec, 0);
    gemm(t1b, Wd2b, dltf, bd2,     Mr, Ec, DTc, 2);

    // 5) Bt, Ct (f32, N=16)
    gemm(ub, WBb, Btf, nullptr, Mr, Nc, Ec, 1);
    gemm(ub, WCb, Ctf, nullptr, Mr, Nc, Ec, 1);

    // 6) selective scan + fused gating -> yb (bf16)
    {
        dim3 grid(Ec / 256, Bc);
        scan_kernel<<<grid, 256, 0, stream>>>(dltf, ub, Btf, Ctf, A_log, W_D,
                                              skipb, ybb, Lc, Ec);
    }

    // 7) out = yb @ W_out^T (f32 -> d_out)
    gemm(ybb, Woub, d_out, nullptr, Mr, Dc, Ec, 1);
}
